// BilstmCrfModel_75058848465518
// MI455X (gfx1250) — matrix operand order repaired and checked by reading, not compiled, not run on hardware
//
#include <hip/hip_runtime.h>
#include <hip/hip_bf16.h>
#include <math.h>

// ---------------------------------------------------------------------------
// BiLSTM-CRF for MI455X (gfx1250, wave32, WMMA bf16 16x16x32)
//   VOCAB=50000 EMB=128 HID=256 (H2=128) L=9 B=128 T=512, gates G=4*H2=512
//
// Pipeline:
//   1) embed:   gather + f32->bf16                         (B,T,128) bf16
//   2) wprep:   Wih/Whh f32 -> bf16
//   3) pregemm: pregate = x @ Wih^T + (bih+bhh)  [WMMA]    (B,T,512) bf16 x2dir
//   4) lstm:    gates = pregate + h @ Whh^T [WMMA, Whh register-stationary],
//               pregate stream double-buffered through registers+LDS so the
//               global-load latency overlaps the whole step body
//   5) emis:    (B*T,256) @ Wout^T(256,9)  (N=9: scalar)
//   6) crf:     numerator + forward logZ, 1 thread / batch row, block-reduce
// ---------------------------------------------------------------------------

typedef __attribute__((ext_vector_type(16))) __bf16 v16bf;
typedef __attribute__((ext_vector_type(8)))  float  v8f;

#define BB   128
#define TT   512
#define EMB  128
#define H2   128
#define GG   512          // 4*H2
#define LL   9

__device__ __forceinline__ float sigf(float x) {
    return 1.0f / (1.0f + __expf(-x));
}

// A fragment (16x32 bf16, ISA 7.12.2 layout):
//   lane L: M = L%16 ; half = L/16 ; elem e: K = kbase + (e/8)*16 + half*8 + (e%8)
__device__ __forceinline__ v16bf load_a_frag(const __bf16* src, int stride, int kbase, int lane) {
    const int m = lane & 15, half = lane >> 4;
    v16bf a;
#pragma unroll
    for (int e = 0; e < 16; ++e) {
        const int k = kbase + ((e >> 3) << 4) + (half << 3) + (e & 7);
        a[e] = src[m * stride + k];
    }
    return a;
}

// B fragment (32x16 bf16): lane L: N = L%16 ; half = L/16 ; elem e: K = kbase + (e/8)*16 + half*8 + (e%8)
// Source W is (G x Kdim) row-major bf16; B[k][n] = W[gbase+n][k]
__device__ __forceinline__ v16bf load_b_frag(const __bf16* w, int gbase, int kbase,
                                             int lane, int kdim) {
    const int n = lane & 15, half = lane >> 4;
    v16bf b;
#pragma unroll
    for (int e = 0; e < 16; ++e) {
        const int k = kbase + ((e >> 3) << 4) + (half << 3) + (e & 7);
        b[e] = w[(size_t)(gbase + n) * kdim + k];
    }
    return b;
}

// ---------------------------------------------------------------------------
// Kernel 1: embedding gather -> bf16, layout (B,T,EMB). 32 rows per block.
// ---------------------------------------------------------------------------
__global__ __launch_bounds__(256) void embed_kernel(const int* __restrict__ seq,
                                                    const float* __restrict__ emb,
                                                    __bf16* __restrict__ xout) {
    const int r0 = blockIdx.x * 32;                   // 2048 blocks * 32 rows
    for (int i = threadIdx.x; i < 32 * EMB; i += 256) {
        const int m = i >> 7, e = i & 127;
        const int tok = seq[r0 + m];
        xout[(size_t)(r0 + m) * EMB + e] = (__bf16)emb[(size_t)tok * EMB + e];
    }
}

// ---------------------------------------------------------------------------
// Kernel 2: weights f32 -> bf16 (wih: 2 x (512x128), whh: 2 x (512x128))
// ---------------------------------------------------------------------------
__global__ __launch_bounds__(256) void wprep_kernel(const float* __restrict__ Wih_f,
                                                    const float* __restrict__ Whh_f,
                                                    const float* __restrict__ Wih_b,
                                                    const float* __restrict__ Whh_b,
                                                    __bf16* __restrict__ wih,
                                                    __bf16* __restrict__ whh) {
    const int i = blockIdx.x * 256 + threadIdx.x;     // 0 .. 262143
    const int half = GG * EMB;                        // 65536 per dir
    if (i < 2 * half) {
        const int dir = i / half, r = i % half;
        wih[i] = (__bf16)((dir ? Wih_b : Wih_f)[r]);
    } else {
        const int j = i - 2 * half;
        const int dir = j / half, r = j % half;
        whh[j] = (__bf16)((dir ? Whh_b : Whh_f)[r]);
    }
}

// ---------------------------------------------------------------------------
// Kernel 3: pregates = x @ Wih^T + (bih+bhh), both directions.  WMMA GEMM,
// M=B*T=65536 (16 rows/block), N=512 (4 tiles/wave), K=128 (4 k-slices).
// ---------------------------------------------------------------------------
__global__ __launch_bounds__(256) void pregemm_kernel(const __bf16* __restrict__ x,
                                                      const __bf16* __restrict__ wih_all,
                                                      const float* __restrict__ bih_f,
                                                      const float* __restrict__ bhh_f,
                                                      const float* __restrict__ bih_b,
                                                      const float* __restrict__ bhh_b,
                                                      __bf16* __restrict__ preg_all) {
    const int blk = blockIdx.x;
    const int dir = blk >> 12;                        // 4096 blocks per dir
    const int m0  = (blk & 4095) * 16;
    const __bf16* w   = wih_all + (size_t)dir * GG * EMB;
    const float*  bih = dir ? bih_b : bih_f;
    const float*  bhh = dir ? bhh_b : bhh_f;
    __bf16* preg = preg_all + (size_t)dir * BB * TT * GG;

    __shared__ __bf16 xs[16 * EMB];                   // 4 KB
    const int tid  = threadIdx.x;
    const int lane = tid & 31;
    const int wave = tid >> 5;

    for (int i = tid; i < 16 * EMB; i += 256)         // rows are contiguous
        xs[i] = x[(size_t)m0 * EMB + i];
    __syncthreads();

    v16bf afr[4];
#pragma unroll
    for (int kk = 0; kk < 4; ++kk) afr[kk] = load_a_frag(xs, EMB, kk * 32, lane);

#pragma unroll
    for (int j = 0; j < 4; ++j) {
        const int gbase = (wave * 4 + j) * 16;
        const float bias = bih[gbase + (lane & 15)] + bhh[gbase + (lane & 15)];
        v8f acc;
#pragma unroll
        for (int r = 0; r < 8; ++r) acc[r] = bias;
#pragma unroll
        for (int kk = 0; kk < 4; ++kk)
            acc = __builtin_amdgcn_wmma_f32_16x16x32_bf16(
                false, afr[kk], false, load_b_frag(w, gbase, kk * 32, lane, EMB),
                (short)0, acc, false, false);
        const int n     = gbase + (lane & 15);
        const int mbase = (lane >> 4) * 8;
#pragma unroll
        for (int r = 0; r < 8; ++r)
            preg[(size_t)(m0 + mbase + r) * GG + n] = (__bf16)acc[r];
    }
}

// ---------------------------------------------------------------------------
// Kernel 4: recurrence.  16 blocks: block = (dir<<3)|wg, 16 batch rows each.
// 256 thr = 8 waves; wave w owns gate N-tiles 4w..4w+3.  Per step:
//   gates(16x512) = pregate_tile + h(16x128) @ Whh^T   (16 WMMAs/wave)
// Whh fragments register-resident (16 v16bf/lane, launch_bounds(256,2)).
// Pregate tile for step t+1 is prefetched into 4 uint4/thread at the top of
// step t and committed to LDS at the end -> global latency overlaps the step.
// LDS: the 32KB f32 `gates` scratch aliases two 16KB bf16 stage buffers
// (used at disjoint phases of each step).
// ---------------------------------------------------------------------------
__global__ __launch_bounds__(256, 2) void lstm_kernel(const __bf16* __restrict__ preg_all,
                                                      const __bf16* __restrict__ whh_all,
                                                      float* __restrict__ hbuf) {
    const int blk = blockIdx.x;
    const int dir = blk >> 3;
    const int b0  = (blk & 7) * 16;
    const __bf16* preg = preg_all + (size_t)dir * BB * TT * GG;
    const __bf16* w    = whh_all + (size_t)dir * GG * H2;

    __shared__ __align__(16) float gates[16 * GG];   // 32 KB (aliased below)
    __shared__ float  cs[16 * H2];                   //  8 KB
    __shared__ __bf16 hs[16 * H2];                   //  4 KB
    __bf16* stage = reinterpret_cast<__bf16*>(gates);   // stage[buf * 8192 + i]

    const int tid  = threadIdx.x;
    const int lane = tid & 31;
    const int wave = tid >> 5;

    for (int i = tid; i < 16 * H2; i += 256) { cs[i] = 0.0f; hs[i] = (__bf16)0.0f; }

    // ---- Whh fragments: resident across all timesteps ----
    v16bf bfrag[4][4];
#pragma unroll
    for (int j = 0; j < 4; ++j)
#pragma unroll
        for (int kk = 0; kk < 4; ++kk)
            bfrag[j][kk] = load_b_frag(w, (wave * 4 + j) * 16, kk * 32, lane, H2);

    // per-thread slice of the 16x512 bf16 tile: 64 bytes = 32 elems
    const int f0 = tid * 32;
    const int pm = f0 >> 9, pg = f0 & 511;

    // prologue: fetch t=0 tile into stage buffer 0
    {
        const int tt0 = dir ? (TT - 1) : 0;
        const uint4* gp = (const uint4*)(preg + ((size_t)(b0 + pm) * TT + tt0) * GG + pg);
        const uint4 a0 = gp[0], a1 = gp[1], a2 = gp[2], a3 = gp[3];
        uint4* sp = (uint4*)(stage + f0);
        sp[0] = a0; sp[1] = a1; sp[2] = a2; sp[3] = a3;
    }
    __syncthreads();

    for (int t = 0; t < TT; ++t) {
        const int tt  = dir ? (TT - 1 - t) : t;
        const int tn  = (t + 1 < TT) ? (t + 1) : t;
        const int ttn = dir ? (TT - 1 - tn) : tn;

        // (1) issue next-step pregate loads; wait lands at commit (6)
        const uint4* gp = (const uint4*)(preg + ((size_t)(b0 + pm) * TT + ttn) * GG + pg);
        const uint4 p0 = gp[0], p1 = gp[1], p2 = gp[2], p3 = gp[3];

        // (2) acc init from current stage buffer (bf16 -> f32)
        const __bf16* st = stage + (size_t)(t & 1) * (16 * GG);
        v8f acc[4];
#pragma unroll
        for (int j = 0; j < 4; ++j) {
            const int n     = (wave * 4 + j) * 16 + (lane & 15);
            const int mbase = (lane >> 4) * 8;
#pragma unroll
            for (int r = 0; r < 8; ++r) acc[j][r] = (float)st[(mbase + r) * GG + n];
        }

        // (3) recurrent WMMAs
#pragma unroll
        for (int kk = 0; kk < 4; ++kk) {
            const v16bf a = load_a_frag(hs, H2, kk * 32, lane);
#pragma unroll
            for (int j = 0; j < 4; ++j)
                acc[j] = __builtin_amdgcn_wmma_f32_16x16x32_bf16(
                    false, a, false, bfrag[j][kk], (short)0, acc[j], false, false);
        }
        __syncthreads();   // all stage reads + hs reads complete

        // (4) scatter f32 gates (overwrites both stage buffers)
#pragma unroll
        for (int j = 0; j < 4; ++j) {
            const int n     = (wave * 4 + j) * 16 + (lane & 15);
            const int mbase = (lane >> 4) * 8;
#pragma unroll
            for (int r = 0; r < 8; ++r)
                gates[(mbase + r) * GG + n] = acc[j][r];
        }
        __syncthreads();

        // (5) elementwise gating: i,f,g,o at N, N+128, N+256, N+384
        for (int i = tid; i < 16 * H2; i += 256) {
            const int m = i >> 7, n = i & 127;
            const float gi = gates[m * GG + n];
            const float gf = gates[m * GG + 128 + n];
            const float gc = gates[m * GG + 256 + n];
            const float go = gates[m * GG + 384 + n];
            const float c  = sigf(gf) * cs[i] + sigf(gi) * tanhf(gc);
            const float h  = sigf(go) * tanhf(c);
            cs[i] = c;
            hs[i] = (__bf16)h;
            hbuf[((size_t)(b0 + m) * TT + tt) * 256 + dir * H2 + n] = h;
        }
        __syncthreads();   // gate reads done; stage area reusable

        // (6) commit prefetched tile into the other stage buffer
        uint4* sp = (uint4*)(stage + (size_t)((t + 1) & 1) * (16 * GG) + f0);
        sp[0] = p0; sp[1] = p1; sp[2] = p2; sp[3] = p3;
        __syncthreads();   // stage ready for next iteration
    }
}

// ---------------------------------------------------------------------------
// Kernel 5: emissions (B*T,256) @ Wout^T (256,9) + bout
// ---------------------------------------------------------------------------
__global__ __launch_bounds__(256) void emis_kernel(const float* __restrict__ hbuf,
                                                   const float* __restrict__ Wout,
                                                   const float* __restrict__ bout,
                                                   float* __restrict__ emis) {
    __shared__ float w[LL * 256];
    __shared__ float bb[LL];
    const int tid = threadIdx.x;
    for (int i = tid; i < LL * 256; i += 256) w[i] = Wout[i];
    if (tid < LL) bb[tid] = bout[tid];
    __syncthreads();

    const int idx = blockIdx.x * 256 + tid;           // 0 .. B*T-1
    if (idx >= BB * TT) return;
    const float* h = hbuf + (size_t)idx * 256;
    float acc[LL];
#pragma unroll
    for (int l = 0; l < LL; ++l) acc[l] = bb[l];
#pragma unroll 4
    for (int k = 0; k < 256; ++k) {
        const float hv = h[k];
#pragma unroll
        for (int l = 0; l < LL; ++l) acc[l] += hv * w[l * 256 + k];
    }
#pragma unroll
    for (int l = 0; l < LL; ++l) emis[(size_t)idx * LL + l] = acc[l];
}

// ---------------------------------------------------------------------------
// Kernel 6: CRF numerator + forward logZ; 1 thread per batch row; reduce.
// ---------------------------------------------------------------------------
__global__ __launch_bounds__(128) void crf_kernel(const int* __restrict__ seq,
                                                  const int* __restrict__ labels,
                                                  const float* __restrict__ emis,
                                                  const float* __restrict__ start_trans,
                                                  const float* __restrict__ end_trans,
                                                  const float* __restrict__ transitions,
                                                  float* __restrict__ out) {
    __shared__ float tr[LL * LL], st[LL], et[LL];
    __shared__ float partial[128];
    const int tid = threadIdx.x;
    if (tid < LL * LL) tr[tid] = transitions[tid];
    if (tid < LL) { st[tid] = start_trans[tid]; et[tid] = end_trans[tid]; }
    __syncthreads();

    const int b = tid;
    const int*   sq = seq    + (size_t)b * TT;
    const int*   lb = labels + (size_t)b * TT;
    const float* em = emis   + (size_t)b * TT * LL;

    float num = st[lb[0]] + em[lb[0]];
    int last = lb[0];
    for (int t = 1; t < TT; ++t) {
        if (sq[t] != 0) {
            num += tr[lb[t - 1] * LL + lb[t]] + em[t * LL + lb[t]];
            last = lb[t];
        }
    }
    num += et[last];

    float alpha[LL];
#pragma unroll
    for (int j = 0; j < LL; ++j) alpha[j] = st[j] + em[j];
    for (int t = 1; t < TT; ++t) {
        if (sq[t] == 0) continue;
        float nxt[LL];
#pragma unroll
        for (int j = 0; j < LL; ++j) {
            float mx = -3.4e38f;
#pragma unroll
            for (int i = 0; i < LL; ++i) mx = fmaxf(mx, alpha[i] + tr[i * LL + j]);
            float s = 0.0f;
#pragma unroll
            for (int i = 0; i < LL; ++i) s += __expf(alpha[i] + tr[i * LL + j] - mx);
            nxt[j] = mx + __logf(s) + em[t * LL + j];
        }
#pragma unroll
        for (int j = 0; j < LL; ++j) alpha[j] = nxt[j];
    }
    float mx = -3.4e38f;
#pragma unroll
    for (int j = 0; j < LL; ++j) mx = fmaxf(mx, alpha[j] + et[j]);
    float s = 0.0f;
#pragma unroll
    for (int j = 0; j < LL; ++j) s += __expf(alpha[j] + et[j] - mx);
    const float logZ = mx + __logf(s);

    partial[tid] = num - logZ;
    __syncthreads();
    for (int off = 64; off > 0; off >>= 1) {
        if (tid < off) partial[tid] += partial[tid + off];
        __syncthreads();
    }
    if (tid == 0) out[0] = -partial[0];
}

// ---------------------------------------------------------------------------
// Launch
// ---------------------------------------------------------------------------
extern "C" void kernel_launch(void* const* d_in, const int* in_sizes, int n_in,
                              void* d_out, int out_size, void* d_ws, size_t ws_size,
                              hipStream_t stream) {
    const int*   seq         = (const int*)d_in[0];
    const int*   labels      = (const int*)d_in[1];
    const float* emb         = (const float*)d_in[2];
    const float* Wih_f       = (const float*)d_in[3];
    const float* Whh_f       = (const float*)d_in[4];
    const float* bih_f       = (const float*)d_in[5];
    const float* bhh_f       = (const float*)d_in[6];
    const float* Wih_b       = (const float*)d_in[7];
    const float* Whh_b       = (const float*)d_in[8];
    const float* bih_b       = (const float*)d_in[9];
    const float* bhh_b       = (const float*)d_in[10];
    const float* Wout        = (const float*)d_in[11];
    const float* bout        = (const float*)d_in[12];
    const float* start_trans = (const float*)d_in[13];
    const float* end_trans   = (const float*)d_in[14];
    const float* transitions = (const float*)d_in[15];
    float* out = (float*)d_out;

    // workspace layout
    char* ws = (char*)d_ws;
    const size_t x_bytes    = (size_t)BB * TT * EMB * sizeof(__bf16);     // 16 MB
    const size_t wih_bytes  = (size_t)2 * GG * EMB * sizeof(__bf16);      // 256 KB
    const size_t whh_bytes  = (size_t)2 * GG * H2 * sizeof(__bf16);       // 256 KB
    const size_t preg_bytes = (size_t)2 * BB * TT * GG * sizeof(__bf16);  // 134 MB
    const size_t hbuf_bytes = (size_t)BB * TT * 256 * sizeof(float);      // 67 MB

    __bf16* x_bf = (__bf16*)ws;
    __bf16* wih  = (__bf16*)(ws + x_bytes);
    __bf16* whh  = (__bf16*)(ws + x_bytes + wih_bytes);
    __bf16* preg = (__bf16*)(ws + x_bytes + wih_bytes + whh_bytes);
    float*  hbuf = (float*)(ws + x_bytes + wih_bytes + whh_bytes + preg_bytes);
    float*  emis = (float*)(ws + x_bytes + wih_bytes + whh_bytes + preg_bytes + hbuf_bytes);

    embed_kernel<<<(BB * TT) / 32, 256, 0, stream>>>(seq, emb, x_bf);
    wprep_kernel<<<(4 * GG * EMB) / 256, 256, 0, stream>>>(Wih_f, Whh_f, Wih_b, Whh_b, wih, whh);
    pregemm_kernel<<<2 * (BB * TT) / 16, 256, 0, stream>>>(x_bf, wih, bih_f, bhh_f, bih_b, bhh_b, preg);
    lstm_kernel<<<16, 256, 0, stream>>>(preg, whh, hbuf);
    emis_kernel<<<(BB * TT) / 256, 256, 0, stream>>>(hbuf, Wout, bout, emis);
    crf_kernel<<<1, 128, 0, stream>>>(seq, labels, emis, start_trans, end_trans, transitions, out);
}
